// MoELayer_515396075686
// MI455X (gfx1250) — matrix operand order repaired and checked by reading, not compiled, not run on hardware
//
#include <hip/hip_runtime.h>
#include <hip/hip_bf16.h>

// ---------------------------------------------------------------------------
// Problem constants (from reference)
// ---------------------------------------------------------------------------
constexpr int N_    = 8192;
constexpr int IN_   = 1024;
constexpr int HID_  = 2048;
constexpr int E_    = 4;
constexpr int TED_  = 100;
constexpr float MU_  = 0.01f;
constexpr float EPS_ = 1e-06f;

constexpr int BM = 128, BN = 128, BK = 32;
constexpr int LDK = BK + 8;                          // 40 halfs (80B row, 16B aligned)
constexpr int BUF_HALFS = (BM + BN) * LDK;           // 10240 halfs / buffer
constexpr int BUF_BYTES = BUF_HALFS * 2;             // 20480 B

typedef __bf16 v16bf __attribute__((ext_vector_type(16)));
typedef float  v8f   __attribute__((ext_vector_type(8)));

union Frag { v16bf bf; uint4 q[2]; };

__device__ __forceinline__ unsigned short f2bf(float f) {
  unsigned u = __builtin_bit_cast(unsigned, f);
  unsigned r = u + 0x7FFFu + ((u >> 16) & 1u);       // round-to-nearest-even
  return (unsigned short)(r >> 16);
}
__device__ __forceinline__ float bf2f(unsigned short h) {
  unsigned u = ((unsigned)h) << 16;
  return __builtin_bit_cast(float, u);
}

// Async global->LDS 16B copy (gfx1250, ASYNCcnt-tracked). saddr form:
//   LDS dest from lds_off (vdst); global src = base (saddr) + byte_off (vaddr)
__device__ __forceinline__ void async_copy_b128(unsigned lds_off, const void* base,
                                                unsigned byte_off) {
  asm volatile("global_load_async_to_lds_b128 %0, %1, %2"
               :: "v"(lds_off), "v"(byte_off), "s"(base)
               : "memory");
}
__device__ __forceinline__ void wait_async_le4() {
  asm volatile("s_wait_asynccnt 0x4" ::: "memory");
}
__device__ __forceinline__ void wait_async_0() {
  asm volatile("s_wait_asynccnt 0x0" ::: "memory");
}

// ---------------------------------------------------------------------------
// Pre-pass 1: X f32 -> bf16 (same layout). 8 elements / thread.
// ---------------------------------------------------------------------------
__global__ __launch_bounds__(256)
void xcvt_kernel(const float* __restrict__ x, unsigned short* __restrict__ xb) {
  size_t i = ((size_t)blockIdx.x * 256 + threadIdx.x) * 8;
  float4 a = *(const float4*)(x + i);
  float4 b = *(const float4*)(x + i + 4);
  uint4 p;
  p.x = (unsigned)f2bf(a.x) | ((unsigned)f2bf(a.y) << 16);
  p.y = (unsigned)f2bf(a.z) | ((unsigned)f2bf(a.w) << 16);
  p.z = (unsigned)f2bf(b.x) | ((unsigned)f2bf(b.y) << 16);
  p.w = (unsigned)f2bf(b.z) | ((unsigned)f2bf(b.w) << 16);
  *(uint4*)(xb + i) = p;
}

// ---------------------------------------------------------------------------
// Pre-pass 2: W[e][K][Nc] f32 -> Wt[e][Nc][K] bf16 (transpose via LDS tile).
// Block = 256 threads handles one 32x32 tile; blockIdx.z = expert.
// ---------------------------------------------------------------------------
__global__ __launch_bounds__(256)
void wt_kernel(const float* __restrict__ W, unsigned short* __restrict__ Wt,
               int K, int Nc) {
  __shared__ unsigned short t[32][40];
  const float* Wi = W + (size_t)blockIdx.z * K * Nc;
  unsigned short* Wo = Wt + (size_t)blockIdx.z * K * Nc;
  const int k0 = blockIdx.y * 32, n0 = blockIdx.x * 32;
  const int tid = threadIdx.x;

  int r  = tid >> 3;             // k-row within tile
  int c4 = (tid & 7) * 4;        // n-col (4 floats)
  float4 f = *(const float4*)(Wi + (size_t)(k0 + r) * Nc + n0 + c4);
  t[c4 + 0][r] = f2bf(f.x);
  t[c4 + 1][r] = f2bf(f.y);
  t[c4 + 2][r] = f2bf(f.z);
  t[c4 + 3][r] = f2bf(f.w);
  __syncthreads();

  int nr = tid >> 3;             // n-row of output
  int kc = (tid & 7) * 4;        // k offset (4 halfs = 8B)
  *(uint2*)(Wo + (size_t)(n0 + nr) * K + k0 + kc) = *(const uint2*)&t[nr][kc];
}

// ---------------------------------------------------------------------------
// Router: one wave32 per sample. logits = [x | temb[task]] @ R  (1124 x 4)
// ---------------------------------------------------------------------------
__global__ __launch_bounds__(256)
void routing_kernel(const float* __restrict__ x, const int* __restrict__ task,
                    const float* __restrict__ temb, const float* __restrict__ R,
                    float* __restrict__ gates, int* __restrict__ idxs,
                    float* __restrict__ reg) {
  int gid  = blockIdx.x * blockDim.x + threadIdx.x;
  int wid  = gid >> 5;
  int lane = threadIdx.x & 31;
  if (wid >= N_) return;

  const float* xr = x + (size_t)wid * IN_;
  const float* te = temb + (size_t)task[wid] * TED_;

  float a0 = 0.f, a1 = 0.f, a2 = 0.f, a3 = 0.f;
  for (int i = lane; i < IN_ + TED_; i += 32) {
    float v = (i < IN_) ? xr[i] : te[i - IN_];
    float4 r = *(const float4*)(R + (size_t)i * 4);
    a0 += v * r.x; a1 += v * r.y; a2 += v * r.z; a3 += v * r.w;
  }
  #pragma unroll
  for (int o = 16; o > 0; o >>= 1) {
    a0 += __shfl_xor(a0, o, 32);
    a1 += __shfl_xor(a1, o, 32);
    a2 += __shfl_xor(a2, o, 32);
    a3 += __shfl_xor(a3, o, 32);
  }
  if (lane == 0) {
    float mx = fmaxf(fmaxf(a0, a1), fmaxf(a2, a3));
    float e0 = __expf(a0 - mx), e1 = __expf(a1 - mx);
    float e2 = __expf(a2 - mx), e3 = __expf(a3 - mx);
    float s  = e0 + e1 + e2 + e3;
    int   im = 0; float lm = a0;
    if (a1 > lm) { lm = a1; im = 1; }
    if (a2 > lm) { lm = a2; im = 2; }
    if (a3 > lm) { lm = a3; im = 3; }
    float g = __expf(lm - mx) / s;
    gates[wid] = g;
    idxs[wid]  = im;
    reg[wid]   = -(MU_ / (float)E_) * (__logf(g + EPS_) + 3.0f * __logf(EPS_));
  }
}

// ---------------------------------------------------------------------------
// bf16 WMMA GEMM + bias + ReLU, async-copy staged, double-buffered LDS.
//   out[m][n] = relu( A[m][:] @ Bt[n][:] + bias[n] )   (Bt is [Nc][K], k-major)
// Tile 128x128x32; 256 threads = 8 waves; each wave: 64x32 (4x2 WMMA tiles).
// ---------------------------------------------------------------------------
__global__ __launch_bounds__(256)
void gemm_bf16_wmma(const unsigned short* __restrict__ A,
                    const unsigned short* __restrict__ Bt,
                    const float* __restrict__ bias,
                    unsigned short* __restrict__ outp,
                    int K, int ldA, int ldOut) {
  __shared__ __align__(16) unsigned short smem[2][BUF_HALFS];

  const int tid  = threadIdx.x;
  const int lane = tid & 31;
  const int wv   = tid >> 5;
  const int m0   = blockIdx.y * BM;
  const int n0   = blockIdx.x * BN;
  const int wm   = (wv & 1) * 64;
  const int wn   = (wv >> 1) * 32;

  const unsigned sbase = (unsigned)(size_t)&smem[0][0];   // LDS byte offset

  // Each thread stages 2x16B of A and 2x16B of B per tile (4 async ops).
  // chunk: idx = tid + i*256 ; row = idx>>2 ; c8 = (idx&3)*8 halfs
  auto stage = [&](int k0, int buf) {
    const unsigned bb = sbase + (unsigned)buf * BUF_BYTES;
    #pragma unroll
    for (int i = 0; i < 2; i++) {
      int idx = tid + i * 256;
      int row = idx >> 2;
      int c8  = (idx & 3) * 8;
      unsigned goffA = (unsigned)(((size_t)(m0 + row) * ldA + k0 + c8) * 2);
      unsigned loffA = bb + (unsigned)((row * LDK + c8) * 2);
      async_copy_b128(loffA, A, goffA);
      unsigned goffB = (unsigned)(((size_t)(n0 + row) * K + k0 + c8) * 2);
      unsigned loffB = bb + (unsigned)((BM * LDK + row * LDK + c8) * 2);
      async_copy_b128(loffB, Bt, goffB);
    }
  };

  v8f acc[4][2] = {};
  const int nsteps = K / BK;

  stage(0, 0);
  for (int s = 0; s < nsteps; ++s) {
    const int cur = s & 1;
    if (s + 1 < nsteps) {
      stage((s + 1) * BK, cur ^ 1);
      wait_async_le4();          // older 4 (tile s) complete (in-order)
    } else {
      wait_async_0();
    }
    __syncthreads();             // all waves' tile-s data visible

    const unsigned short* sa = &smem[cur][0];
    const unsigned short* sb = &smem[cur][BM * LDK];

    // A 16x32 frag: lane<16 -> M=lane, K={0..7,16..23}; lane>=16 -> M=lane-16, K={8..15,24..31}
    Frag a[4], b[2];
    const int kb = (lane >> 4) * 8;
    #pragma unroll
    for (int mi = 0; mi < 4; mi++) {
      int row = wm + mi * 16 + (lane & 15);
      a[mi].q[0] = *(const uint4*)(sa + row * LDK + kb);
      a[mi].q[1] = *(const uint4*)(sa + row * LDK + kb + 16);
    }
    // B 32x16 frag: lane<16 -> N=lane, K={0..7,16..23} ; lane>=16 -> N=lane-16, K={8..15,24..31}
    const int kb2 = (lane >> 4) * 8;
    #pragma unroll
    for (int ni = 0; ni < 2; ni++) {
      int col = wn + ni * 16 + (lane & 15);
      b[ni].q[0] = *(const uint4*)(sb + col * LDK + kb2);
      b[ni].q[1] = *(const uint4*)(sb + col * LDK + kb2 + 16);
    }
    #pragma unroll
    for (int mi = 0; mi < 4; mi++)
      #pragma unroll
      for (int ni = 0; ni < 2; ni++)
        acc[mi][ni] = __builtin_amdgcn_wmma_f32_16x16x32_bf16(
            false, a[mi].bf, false, b[ni].bf,
            (short)0, acc[mi][ni], false, false);

    __syncthreads();             // everyone done reading buf cur before reuse
  }

  // Epilogue: bias + ReLU, bf16 store.
  // C 16x16 f32: VGPR v, lane<16 -> (M=v, N=lane); lane>=16 -> (M=v+8, N=lane-16)
  #pragma unroll
  for (int mi = 0; mi < 4; mi++) {
    #pragma unroll
    for (int ni = 0; ni < 2; ni++) {
      int col  = n0 + wn + ni * 16 + (lane & 15);
      float bv = bias[col];
      int rbase = m0 + wm + mi * 16 + (lane >> 4) * 8;
      #pragma unroll
      for (int v = 0; v < 8; v++) {
        float val = acc[mi][ni][v] + bv;
        val = val > 0.f ? val : 0.f;
        outp[(size_t)(rbase + v) * ldOut + col] = f2bf(val);
      }
    }
  }
}

// ---------------------------------------------------------------------------
// Per-sample classical Gram-Schmidt over 4 expert vectors (dim 2048),
// then tower = gate * ortho[idx]. One 256-thread block per sample.
// ---------------------------------------------------------------------------
__device__ __forceinline__ float block_reduce_sum(float v, float* red) {
  int t = threadIdx.x;
  #pragma unroll
  for (int o = 16; o > 0; o >>= 1) v += __shfl_down(v, o, 32);
  if ((t & 31) == 0) red[t >> 5] = v;
  __syncthreads();
  if (t == 0) {
    float s = 0.f;
    #pragma unroll
    for (int i = 0; i < 8; i++) s += red[i];
    red[0] = s;
  }
  __syncthreads();
  float r = red[0];
  __syncthreads();
  return r;
}

__global__ __launch_bounds__(256)
void gs_kernel(const unsigned short* __restrict__ eo,
               const float* __restrict__ gates, const int* __restrict__ idxs,
               float* __restrict__ tower) {
  __shared__ float red[8];
  const int n = blockIdx.x;
  const int t = threadIdx.x;
  const unsigned short* base = eo + (size_t)n * E_ * HID_;

  float v[4][8];
  #pragma unroll
  for (int e = 0; e < 4; e++) {
    uint4 q = *(const uint4*)(base + e * HID_ + t * 8);
    const unsigned short* u = (const unsigned short*)&q;
    #pragma unroll
    for (int j = 0; j < 8; j++) v[e][j] = bf2f(u[j]);
  }

  float b[4][8];
  {
    float s = 0.f;
    #pragma unroll
    for (int j = 0; j < 8; j++) s += v[0][j] * v[0][j];
    float nrm = block_reduce_sum(s, red);
    float inv = __frsqrt_rn(nrm + 1e-30f);
    #pragma unroll
    for (int j = 0; j < 8; j++) b[0][j] = v[0][j] * inv;
  }
  #pragma unroll
  for (int i = 1; i < 4; i++) {
    float coef[3];
    for (int jj = 0; jj < i; jj++) {
      float s = 0.f;
      #pragma unroll
      for (int j = 0; j < 8; j++) s += v[i][j] * b[jj][j];
      coef[jj] = block_reduce_sum(s, red);
    }
    float w[8];
    #pragma unroll
    for (int j = 0; j < 8; j++) {
      float acc = v[i][j];
      for (int jj = 0; jj < i; jj++) acc -= coef[jj] * b[jj][j];
      w[j] = acc;
    }
    float s = 0.f;
    #pragma unroll
    for (int j = 0; j < 8; j++) s += w[j] * w[j];
    float nrm = block_reduce_sum(s, red);
    float inv = __frsqrt_rn(nrm + 1e-30f);
    #pragma unroll
    for (int j = 0; j < 8; j++) b[i][j] = w[j] * inv;
  }

  const int   id = idxs[n];
  const float g  = gates[n];
  float o[8] = {0.f, 0.f, 0.f, 0.f, 0.f, 0.f, 0.f, 0.f};
  #pragma unroll
  for (int e = 0; e < 4; e++) {
    float m = (e == id) ? g : 0.f;
    #pragma unroll
    for (int j = 0; j < 8; j++) o[j] += m * b[e][j];
  }
  float* dst = tower + (size_t)n * HID_ + t * 8;
  #pragma unroll
  for (int j = 0; j < 8; j++) dst[j] = o[j];
}

// ---------------------------------------------------------------------------
// Launcher
// ---------------------------------------------------------------------------
extern "C" void kernel_launch(void* const* d_in, const int* in_sizes, int n_in,
                              void* d_out, int out_size, void* d_ws, size_t ws_size,
                              hipStream_t stream) {
  const float* x    = (const float*)d_in[0];
  const int*   task = (const int*)  d_in[1];
  const float* temb = (const float*)d_in[2];
  const float* R    = (const float*)d_in[3];
  const float* W1   = (const float*)d_in[4];
  const float* b1   = (const float*)d_in[5];
  const float* W2   = (const float*)d_in[6];
  const float* b2   = (const float*)d_in[7];

  float* tower = (float*)d_out;
  float* reg   = tower + (size_t)N_ * HID_;

  // ws: xb | W1t | W2t | h | eo | gates | idx   (all bf16 except last two)
  char* ws = (char*)d_ws;
  size_t off = 0;
  unsigned short* xb   = (unsigned short*)(ws + off); off += (size_t)N_ * IN_ * 2;
  unsigned short* w1t  = (unsigned short*)(ws + off); off += (size_t)E_ * IN_ * HID_ * 2;
  unsigned short* w2t  = (unsigned short*)(ws + off); off += (size_t)E_ * HID_ * HID_ * 2;
  unsigned short* h_ws = (unsigned short*)(ws + off); off += (size_t)N_ * HID_ * 2;
  unsigned short* eo_ws= (unsigned short*)(ws + off); off += (size_t)N_ * E_ * HID_ * 2;
  float* gates = (float*)(ws + off); off += (size_t)N_ * sizeof(float);
  int*   idxs  = (int*)  (ws + off);

  // pre-passes: bf16 conversion (X) and conversion+transpose (W1, W2)
  xcvt_kernel<<<(N_ * IN_) / (256 * 8), 256, 0, stream>>>(x, xb);
  wt_kernel<<<dim3(HID_ / 32, IN_ / 32, E_), 256, 0, stream>>>(W1, w1t, IN_, HID_);
  wt_kernel<<<dim3(HID_ / 32, HID_ / 32, E_), 256, 0, stream>>>(W2, w2t, HID_, HID_);

  routing_kernel<<<(N_ * 32) / 256, 256, 0, stream>>>(x, task, temb, R, gates, idxs, reg);

  dim3 grid(HID_ / BN, N_ / BM);
  for (int e = 0; e < E_; e++) {
    gemm_bf16_wmma<<<grid, 256, 0, stream>>>(
        xb, w1t + (size_t)e * IN_ * HID_, b1 + (size_t)e * HID_,
        h_ws, IN_, IN_, HID_);
    gemm_bf16_wmma<<<grid, 256, 0, stream>>>(
        h_ws, w2t + (size_t)e * HID_ * HID_, b2 + (size_t)e * HID_,
        eo_ws + (size_t)e * HID_, HID_, HID_, E_ * HID_);
  }

  gs_kernel<<<N_, 256, 0, stream>>>(eo_ws, gates, idxs, tower);
}
